// FlowSurrogate_32384053411842
// MI455X (gfx1250) — matrix operand order repaired and checked by reading, not compiled, not run on hardware
//
#include <hip/hip_runtime.h>
#include <cstddef>

// ---------------------------------------------------------------------------
// FlowSurrogate for MI455X (gfx1250).
// Compute-bound (~1.6 TFLOP vs ~35MB footprint) -> f16 WMMA with fp32 accum.
// One fused kernel runs the full 25-step recurrence with activations in LDS.
// 32 batch rows per workgroup (2 M-tiles per wave) so each B-fragment fetched
// from L2 feeds two WMMAs -> halves device-level weight streaming (~50GB L2).
// Epilogue uses the CDNA5 hardware V_TANH_F32 (branch-free).
// ---------------------------------------------------------------------------

typedef __attribute__((ext_vector_type(16))) _Float16 v16h;
typedef __attribute__((ext_vector_type(8)))  _Float16 v8h;
typedef __attribute__((ext_vector_type(8)))  float    v8f;

// ---- problem dims ----
constexpr int BTOT = 65536;
constexpr int SEQ4 = 104;   // output width
constexpr int ROWS = 32;    // batch rows per workgroup (2 WMMA M-tiles)
// ---- padded dims (N mult of 16, K mult of 32) ----
constexpr int PRE_N = 528;  // 520 -> 33 n-chunks
constexpr int PRE_K = 544;  // 520 -> 17 k-chunks
constexpr int S_N   = 112;  // 104 -> 7 n-chunks
constexpr int S_K   = 128;  // 104 -> 4 k-chunks
constexpr int H_N   = 208;  // 200 -> 13 n-chunks
constexpr int H_K   = 224;  // 200 -> 7 k-chunks
constexpr int IN_K  = 32;   // 8   -> 1 k-chunk
constexpr int O_N   = 16;   // 4   -> 1 n-chunk
constexpr int DP_K  = 128;  // dp first layer K (<=100) -> 4 k-chunks

// LDS partition (halves): yb, d1a, d1b, d1s, xa, xb, xin
constexpr int LDS_HALVES = ROWS * (DP_K + PRE_K + PRE_K + S_K + H_K + H_K + IN_K);
constexpr size_t LDS_BYTES = (size_t)LDS_HALVES * 2;  // 116,736 B (<320KB/WGP)

// ---- f16 workspace layout (element offsets) ----
constexpr size_t OFF_DPW  = 0;                                    // [25][528][128]
constexpr size_t OFF_DP1W = OFF_DPW  + (size_t)25 * PRE_N * DP_K; // [528][544]
constexpr size_t OFF_DP2W = OFF_DP1W + (size_t)PRE_N * PRE_K;     // [112][544]
constexpr size_t OFF_L1W  = OFF_DP2W + (size_t)S_N * PRE_K;       // [208][32]
constexpr size_t OFF_L1RW = OFF_L1W  + (size_t)H_N * IN_K;        // [208][128]
constexpr size_t OFF_L2W  = OFF_L1RW + (size_t)H_N * S_K;         // [208][224]
constexpr size_t OFF_L2RW = OFF_L2W  + (size_t)H_N * H_K;         // [208][128]
constexpr size_t OFF_L3W  = OFF_L2RW + (size_t)H_N * S_K;         // [208][224]
constexpr size_t OFF_L3RW = OFF_L3W  + (size_t)H_N * H_K;         // [208][128]
constexpr size_t OFF_L4W  = OFF_L3RW + (size_t)H_N * S_K;         // [16][224]
constexpr size_t OFF_L4RW = OFF_L4W  + (size_t)O_N * H_K;         // [16][128]
constexpr size_t OFF_XH   = OFF_L4RW + (size_t)O_N * S_K;         // [B][32] f16 x
constexpr size_t HALF_TOTAL = OFF_XH + (size_t)BTOT * IN_K;
constexpr size_t HALF_BYTES = HALF_TOTAL * 2;                     // ~8.6 MB

// ---- f32 bias workspace (element offsets, after the f16 region) ----
constexpr size_t F_DPB  = 0;                    // [25][528]
constexpr size_t F_DP1B = F_DPB  + 25 * PRE_N;  // [528]
constexpr size_t F_DP2B = F_DP1B + PRE_N;       // [112]
constexpr size_t F_B1   = F_DP2B + S_N;         // [208] = l1_b + l1r_b
constexpr size_t F_B2   = F_B1   + H_N;         // [208]
constexpr size_t F_B3   = F_B2   + H_N;         // [208]
constexpr size_t F_B4   = F_B3   + H_N;         // [16]

// ---------------------------------------------------------------------------
// Branch-free tanh: CDNA5 hardware V_TANH_F32 (confirmed in disasm).
// ---------------------------------------------------------------------------
__device__ __forceinline__ float fast_tanh(float x) {
#if __has_builtin(__builtin_amdgcn_tanhf)
  return __builtin_amdgcn_tanhf(x);
#else
  float e = __builtin_amdgcn_exp2f(x * 2.8853900817779268f);  // 2*log2(e)
  float r = __builtin_amdgcn_rcpf(e + 1.0f);
  return __builtin_fmaf(-2.0f, r, 1.0f);
#endif
}

// ===========================================================================
// Prep kernels: pad/convert weights to f16, combine+pad biases, convert x.
// ===========================================================================
__global__ void pad_w_kernel(const float* __restrict__ src, _Float16* __restrict__ dst,
                             int N, int K, int Npad, int Kpad, int Klim) {
  int idx = blockIdx.x * blockDim.x + threadIdx.x;
  if (idx >= Npad * Kpad) return;
  int n = idx / Kpad, k = idx % Kpad;
  float v = (n < N && k < K && k < Klim) ? src[(size_t)n * K + k] : 0.0f;
  dst[idx] = (_Float16)v;
}

__global__ void pad_dpb_kernel(const float* __restrict__ src, float* __restrict__ dst) {
  int idx = blockIdx.x * blockDim.x + threadIdx.x;
  if (idx >= 25 * PRE_N) return;
  int i = idx / PRE_N, n = idx % PRE_N;
  dst[idx] = (n < 520) ? src[(size_t)i * 520 + n] : 0.0f;
}

__global__ void pad_bias_kernel(const float* __restrict__ a, const float* __restrict__ b,
                                float* __restrict__ dst, int N, int Npad) {
  int n = blockIdx.x * blockDim.x + threadIdx.x;
  if (n >= Npad) return;
  float v = 0.0f;
  if (n < N) { v = a[n]; if (b) v += b[n]; }
  dst[n] = v;
}

__global__ void xconv_kernel(const float* __restrict__ x, _Float16* __restrict__ xh) {
  int idx = blockIdx.x * blockDim.x + threadIdx.x;
  if (idx >= BTOT * IN_K) return;
  int r = idx >> 5, c = idx & 31;
  xh[idx] = (_Float16)((c < 8) ? x[(size_t)r * 8 + c] : 0.0f);
}

// ===========================================================================
// WMMA fragment helpers (wave32, v_wmma_f32_16x16x32_f16).
// A (16xK, 16-bit): lanes 0-15 = M, hold K {0..7,16..23}; lanes 16-31 hold
// K {8..15,24..31}.  B (Kx16 from row-major W[N][K]): lane n%16 = column N,
// same K split as A: lanes 0-15 hold K {0..7,16..23}, lanes 16-31 K {8..15,24..31}.
// ===========================================================================
__device__ __forceinline__ v16h load_a_frag(const _Float16* A, int lane, int kc, int kpad) {
  const int m = lane & 15;
  const int koff = (lane < 16) ? 0 : 8;
  const _Float16* base = A + m * kpad + kc * 32;
  v8h lo = *(const v8h*)(base + koff);
  v8h hi = *(const v8h*)(base + 16 + koff);
  v16h r;
#pragma unroll
  for (int i = 0; i < 8; ++i) { r[i] = lo[i]; r[i + 8] = hi[i]; }
  return r;
}

__device__ __forceinline__ v16h load_b_frag(const _Float16* __restrict__ W,
                                            int lane, int nc, int kc, int kpad) {
  const int n = nc * 16 + (lane & 15);
  const int k0 = kc * 32 + ((lane < 16) ? 0 : 8);
  const v8h* p = (const v8h*)(W + (size_t)n * kpad + k0);
  v8h lo = p[0], hi = p[2];
  v16h r;
#pragma unroll
  for (int i = 0; i < 8; ++i) { r[i] = lo[i]; r[i + 8] = hi[i]; }
  return r;
}

// C/D 16x16 f32: lane n%16 = column N; VGPR j = row M j (+8 for lanes 16-31).
__device__ __forceinline__ void store_tile_tanh(_Float16* C, int cpad, int lane, int nc,
                                                v8f acc, float bv) {
  const int n = nc * 16 + (lane & 15);
  const int mb = (lane < 16) ? 0 : 8;
  _Float16* p = C + mb * cpad + n;
#pragma unroll
  for (int j = 0; j < 8; ++j)
    p[j * cpad] = (_Float16)fast_tanh(acc[j] + bv);
}

// C[0:32,:] = tanh(A1 @ W1^T + A2 @ W2^T + bias), two 16-row M-tiles per wave
// sharing every B fragment; N chunks split across the 4 waves.
__device__ __forceinline__ void gemm_tile(
    const _Float16* __restrict__ W1, const _Float16* A1, int kpad1, int kc1,
    const _Float16* __restrict__ W2, const _Float16* A2, int kpad2, int kc2,
    const float* __restrict__ bias, _Float16* C, int cpad,
    int nchunks, int lane, int wave) {
  for (int nc = wave; nc < nchunks; nc += 4) {
    v8f acc0 = {}, acc1 = {};
    for (int kc = 0; kc < kc1; ++kc) {
      v16h b  = load_b_frag(W1, lane, nc, kc, kpad1);
      v16h a0 = load_a_frag(A1, lane, kc, kpad1);
      v16h a1 = load_a_frag(A1 + 16 * kpad1, lane, kc, kpad1);
      acc0 = __builtin_amdgcn_wmma_f32_16x16x32_f16(false, a0, false, b, (short)0, acc0, false, false);
      acc1 = __builtin_amdgcn_wmma_f32_16x16x32_f16(false, a1, false, b, (short)0, acc1, false, false);
    }
    for (int kc = 0; kc < kc2; ++kc) {
      v16h b  = load_b_frag(W2, lane, nc, kc, kpad2);
      v16h a0 = load_a_frag(A2, lane, kc, kpad2);
      v16h a1 = load_a_frag(A2 + 16 * kpad2, lane, kc, kpad2);
      acc0 = __builtin_amdgcn_wmma_f32_16x16x32_f16(false, a0, false, b, (short)0, acc0, false, false);
      acc1 = __builtin_amdgcn_wmma_f32_16x16x32_f16(false, a1, false, b, (short)0, acc1, false, false);
    }
    const float bv = bias[nc * 16 + (lane & 15)];
    store_tile_tanh(C, cpad, lane, nc, acc0, bv);
    store_tile_tanh(C + 16 * cpad, cpad, lane, nc, acc1, bv);
  }
}

// ===========================================================================
// Main fused kernel: 32 batch rows per workgroup, 4 waves, full 25-step
// recurrence with all activations resident in (dynamic) LDS, f16.
// ===========================================================================
__global__ __launch_bounds__(128) void flow_main_kernel(
    const _Float16* __restrict__ wsh, const float* __restrict__ wsf,
    float* __restrict__ out) {
  extern __shared__ __align__(16) _Float16 lds[];
  _Float16* yb  = lds;                    // [32][DP_K]  recurrent y state
  _Float16* d1a = yb  + ROWS * DP_K;      // [32][PRE_K] dp0 out / dp1 in
  _Float16* d1b = d1a + ROWS * PRE_K;     // [32][PRE_K] dp1 out / dp2 in
  _Float16* d1s = d1b + ROWS * PRE_K;     // [32][S_K]   dp2 out ("d1")
  _Float16* xa  = d1s + ROWS * S_K;       // [32][H_K]   hidden ping
  _Float16* xb  = xa  + ROWS * H_K;       // [32][H_K]   hidden pong
  _Float16* xin = xb  + ROWS * H_K;       // [32][IN_K]  f16 x

  const int tid  = threadIdx.x;
  const int lane = tid & 31;
  const int wave = tid >> 5;
  const int row0 = blockIdx.x * ROWS;

  // ---- init LDS (padding columns stay zero for the whole kernel) ----
  for (int i = tid; i < ROWS * DP_K; i += 128) {
    int c = i & (DP_K - 1);
    yb[i] = (_Float16)((c < SEQ4) ? 1.0f : 0.0f);  // y starts as ones
  }
  for (int i = tid; i < ROWS * S_K;   i += 128) d1s[i] = (_Float16)0.0f;
  for (int i = tid; i < ROWS * PRE_K; i += 128) { d1a[i] = (_Float16)0.0f; d1b[i] = (_Float16)0.0f; }
  for (int i = tid; i < ROWS * H_K;   i += 128) { xa[i] = (_Float16)0.0f;  xb[i] = (_Float16)0.0f; }
  for (int i = tid; i < ROWS * IN_K;  i += 128) xin[i] = wsh[OFF_XH + (size_t)row0 * IN_K + i];
  {  // y columns 0..3 are never written by the loop -> stay 1.0 in the output
    int r = tid >> 2, c = tid & 3;
    out[(size_t)(row0 + r) * SEQ4 + c] = 1.0f;
  }
  __syncthreads();

  for (int it = 1; it <= 25; ++it) {
    // dp0: d1a = tanh(y[:, :it*4] @ dpw[it]^T + dpb[it]); the K-slice is
    // folded into the zero-padded weights, so a fixed 4 k-chunks is exact.
    const _Float16* w0 = wsh + OFF_DPW + (size_t)(it - 1) * PRE_N * DP_K;
    const float*    b0 = wsf + F_DPB + (size_t)(it - 1) * PRE_N;
    gemm_tile(w0, yb, DP_K, 4,  w0, yb, DP_K, 0,  b0, d1a, PRE_K, PRE_N / 16, lane, wave);
    __syncthreads();

    // dp1: d1b = tanh(d1a @ dp1_w^T + dp1_b)
    gemm_tile(wsh + OFF_DP1W, d1a, PRE_K, PRE_K / 32,
              wsh + OFF_DP1W, d1a, PRE_K, 0,
              wsf + F_DP1B, d1b, PRE_K, PRE_N / 16, lane, wave);
    __syncthreads();

    // dp2: d1s = tanh(d1b @ dp2_w^T + dp2_b)
    gemm_tile(wsh + OFF_DP2W, d1b, PRE_K, PRE_K / 32,
              wsh + OFF_DP2W, d1b, PRE_K, 0,
              wsf + F_DP2B, d1s, S_K, S_N / 16, lane, wave);
    __syncthreads();

    // l1: xa = tanh(x @ l1_w^T + d1s @ l1r_w^T + b1)
    gemm_tile(wsh + OFF_L1W,  xin, IN_K, 1,
              wsh + OFF_L1RW, d1s, S_K, S_K / 32,
              wsf + F_B1, xa, H_K, H_N / 16, lane, wave);
    __syncthreads();

    // l2: xb = tanh(xa @ l2_w^T + d1s @ l2r_w^T + b2)
    gemm_tile(wsh + OFF_L2W,  xa,  H_K, H_K / 32,
              wsh + OFF_L2RW, d1s, S_K, S_K / 32,
              wsf + F_B2, xb, H_K, H_N / 16, lane, wave);
    __syncthreads();

    // l3: xa = tanh(xb @ l3_w^T + d1s @ l3r_w^T + b3)
    gemm_tile(wsh + OFF_L3W,  xb,  H_K, H_K / 32,
              wsh + OFF_L3RW, d1s, S_K, S_K / 32,
              wsf + F_B3, xa, H_K, H_N / 16, lane, wave);
    __syncthreads();

    // l4 (no tanh, N=4 in one 16-chunk, wave 0 only, both M-tiles): write
    // output columns it*4..it*4+3 and feed back into the recurrent y buffer.
    if (wave == 0) {
      v8f acc0 = {}, acc1 = {};
      for (int kc = 0; kc < H_K / 32; ++kc) {
        v16h b  = load_b_frag(wsh + OFF_L4W, lane, 0, kc, H_K);
        v16h a0 = load_a_frag(xa, lane, kc, H_K);
        v16h a1 = load_a_frag(xa + 16 * H_K, lane, kc, H_K);
        acc0 = __builtin_amdgcn_wmma_f32_16x16x32_f16(false, a0, false, b, (short)0, acc0, false, false);
        acc1 = __builtin_amdgcn_wmma_f32_16x16x32_f16(false, a1, false, b, (short)0, acc1, false, false);
      }
      for (int kc = 0; kc < S_K / 32; ++kc) {
        v16h b  = load_b_frag(wsh + OFF_L4RW, lane, 0, kc, S_K);
        v16h a0 = load_a_frag(d1s, lane, kc, S_K);
        v16h a1 = load_a_frag(d1s + 16 * S_K, lane, kc, S_K);
        acc0 = __builtin_amdgcn_wmma_f32_16x16x32_f16(false, a0, false, b, (short)0, acc0, false, false);
        acc1 = __builtin_amdgcn_wmma_f32_16x16x32_f16(false, a1, false, b, (short)0, acc1, false, false);
      }
      const int nl = lane & 15;
      const int mb = (lane < 16) ? 0 : 8;
      if (nl < 4) {
        const float bv = wsf[F_B4 + nl];
#pragma unroll
        for (int j = 0; j < 8; ++j) {
          float v0 = acc0[j] + bv;
          float v1 = acc1[j] + bv;
          int m = mb + j;
          out[(size_t)(row0 + m) * SEQ4 + it * 4 + nl] = v0;
          out[(size_t)(row0 + 16 + m) * SEQ4 + it * 4 + nl] = v1;
          yb[m * DP_K + it * 4 + nl] = (_Float16)v0;
          yb[(16 + m) * DP_K + it * 4 + nl] = (_Float16)v1;
        }
      }
    }
    __syncthreads();
  }
}

// ===========================================================================
extern "C" void kernel_launch(void* const* d_in, const int* in_sizes, int n_in,
                              void* d_out, int out_size, void* d_ws, size_t ws_size,
                              hipStream_t stream) {
  const float* x     = (const float*)d_in[0];
  const float* dp_w  = (const float*)d_in[1];
  const float* dp_b  = (const float*)d_in[2];
  const float* dp1_w = (const float*)d_in[3];
  const float* dp1_b = (const float*)d_in[4];
  const float* dp2_w = (const float*)d_in[5];
  const float* dp2_b = (const float*)d_in[6];
  const float* l1_w  = (const float*)d_in[7];
  const float* l1_b  = (const float*)d_in[8];
  const float* l1r_w = (const float*)d_in[9];
  const float* l1r_b = (const float*)d_in[10];
  const float* l2_w  = (const float*)d_in[11];
  const float* l2_b  = (const float*)d_in[12];
  const float* l2r_w = (const float*)d_in[13];
  const float* l2r_b = (const float*)d_in[14];
  const float* l3_w  = (const float*)d_in[15];
  const float* l3_b  = (const float*)d_in[16];
  const float* l3r_w = (const float*)d_in[17];
  const float* l3r_b = (const float*)d_in[18];
  const float* l4_w  = (const float*)d_in[19];
  const float* l4_b  = (const float*)d_in[20];
  const float* l4r_w = (const float*)d_in[21];
  const float* l4r_b = (const float*)d_in[22];

  _Float16* wsh = (_Float16*)d_ws;                       // f16 region (~8.6MB)
  float*    wsf = (float*)((char*)d_ws + HALF_BYTES);    // f32 bias region
  float*    out = (float*)d_out;

  auto nb = [](long n, int b) { return (unsigned)((n + b - 1) / b); };

  // dp_w[i] sliced to columns < (i+1)*4 in the reference -> fold into padding.
  for (int i = 0; i < 25; ++i) {
    pad_w_kernel<<<nb((long)PRE_N * DP_K, 256), 256, 0, stream>>>(
        dp_w + (size_t)i * 520 * 100, wsh + OFF_DPW + (size_t)i * PRE_N * DP_K,
        520, 100, PRE_N, DP_K, (i + 1) * 4);
  }
  pad_w_kernel<<<nb((long)PRE_N * PRE_K, 256), 256, 0, stream>>>(dp1_w, wsh + OFF_DP1W, 520, 520, PRE_N, PRE_K, 520);
  pad_w_kernel<<<nb((long)S_N * PRE_K, 256), 256, 0, stream>>>(dp2_w, wsh + OFF_DP2W, 104, 520, S_N, PRE_K, 520);
  pad_w_kernel<<<nb((long)H_N * IN_K, 256), 256, 0, stream>>>(l1_w,  wsh + OFF_L1W,  200, 8,   H_N, IN_K, 8);
  pad_w_kernel<<<nb((long)H_N * S_K, 256), 256, 0, stream>>>(l1r_w, wsh + OFF_L1RW, 200, 104, H_N, S_K, 104);
  pad_w_kernel<<<nb((long)H_N * H_K, 256), 256, 0, stream>>>(l2_w,  wsh + OFF_L2W,  200, 200, H_N, H_K, 200);
  pad_w_kernel<<<nb((long)H_N * S_K, 256), 256, 0, stream>>>(l2r_w, wsh + OFF_L2RW, 200, 104, H_N, S_K, 104);
  pad_w_kernel<<<nb((long)H_N * H_K, 256), 256, 0, stream>>>(l3_w,  wsh + OFF_L3W,  200, 200, H_N, H_K, 200);
  pad_w_kernel<<<nb((long)H_N * S_K, 256), 256, 0, stream>>>(l3r_w, wsh + OFF_L3RW, 200, 104, H_N, S_K, 104);
  pad_w_kernel<<<nb((long)O_N * H_K, 256), 256, 0, stream>>>(l4_w,  wsh + OFF_L4W,  4,   200, O_N, H_K, 200);
  pad_w_kernel<<<nb((long)O_N * S_K, 256), 256, 0, stream>>>(l4r_w, wsh + OFF_L4RW, 4,   104, O_N, S_K, 104);

  pad_dpb_kernel<<<nb(25L * PRE_N, 256), 256, 0, stream>>>(dp_b, wsf + F_DPB);
  pad_bias_kernel<<<nb(PRE_N, 256), 256, 0, stream>>>(dp1_b, nullptr, wsf + F_DP1B, 520, PRE_N);
  pad_bias_kernel<<<nb(S_N, 256), 256, 0, stream>>>(dp2_b, nullptr, wsf + F_DP2B, 104, S_N);
  pad_bias_kernel<<<nb(H_N, 256), 256, 0, stream>>>(l1_b, l1r_b, wsf + F_B1, 200, H_N);
  pad_bias_kernel<<<nb(H_N, 256), 256, 0, stream>>>(l2_b, l2r_b, wsf + F_B2, 200, H_N);
  pad_bias_kernel<<<nb(H_N, 256), 256, 0, stream>>>(l3_b, l3r_b, wsf + F_B3, 200, H_N);
  pad_bias_kernel<<<nb(O_N, 256), 256, 0, stream>>>(l4_b, l4r_b, wsf + F_B4, 4, O_N);

  xconv_kernel<<<nb((long)BTOT * IN_K, 256), 256, 0, stream>>>(x, wsh + OFF_XH);

  flow_main_kernel<<<BTOT / ROWS, 128, LDS_BYTES, stream>>>(wsh, wsf, out);

  (void)in_sizes; (void)n_in; (void)out_size; (void)ws_size;
}
